// Network_89008902242787
// MI455X (gfx1250) — matrix operand order repaired and checked by reading, not compiled, not run on hardware
//
#include <hip/hip_runtime.h>
#include <hip/hip_bf16.h>

typedef __attribute__((ext_vector_type(16))) _Float16 v16h;
typedef __attribute__((ext_vector_type(8)))  float    v8f;

#define NROWS 8192
#define BIGF  1000000000.0f

// ---------------- WMMA helpers (gfx1250 wave32, 16x16x32 f16 -> f32) ---------

__device__ inline v8f vzero8() {
  v8f r;
#pragma unroll
  for (int i = 0; i < 8; ++i) r[i] = 0.0f;
  return r;
}

__device__ inline v8f wmma32f16(v16h a, v16h b, v8f c) {
  return __builtin_amdgcn_wmma_f32_16x16x32_f16(false, a, false, b, (short)0, c,
                                                false, false);
}

// A-matrix 16x32 f16 fragment (ISA 7.12.2). Per lane the 16 halves are two
// contiguous 8-half runs of the A row: [k0+kb .. +7] and [k0+kb+16 .. +23]
// (kb = 0 for lanes 0-15, 8 for lanes 16-31)  ->  2 x b128 loads.
__device__ inline v16h frag_a(const _Float16* A, int lda, int m0, int k0, int lane) {
  int m  = m0 + (lane & 15);
  int kb = (lane < 16) ? 0 : 8;
  const _Float16* row = A + (size_t)m * lda + k0 + kb;
  union { v16h v; uint4 q[2]; } u;
  u.q[0] = *(const uint4*)(row);
  u.q[1] = *(const uint4*)(row + 16);
  return u.v;
}

// B fragments come from weights pre-swizzled by k_pack into fragment-native
// layout: tile (kt,nt) stores lane-major 32x16 halves contiguously.
// -> 2 x b128 loads, fully coalesced across the wave.
__device__ inline v16h frag_bp(const _Float16* P, int kt, int nt, int ntiles, int lane) {
  const _Float16* base = P + (((size_t)kt * ntiles + nt) * 32 + lane) * 16;
  union { v16h v; uint4 q[2]; } u;
  u.q[0] = *(const uint4*)(base);
  u.q[1] = *(const uint4*)(base + 8);
  return u.v;
}

// ---------------- weight pack: f32 row-major (Kd x Nw) -> f16 fragment layout
// element h of lane in tile (kt,nt):  n = nt*16 + (lane&15),
// k = kt*32 + (lane<16?0:8) + (h&7) + ((h>>3)<<4).  Rows k>=Ksrc are zero
// (used to pad W_t3 from K=48 to K=64).

__global__ void k_pack(const float* __restrict__ W, _Float16* __restrict__ P,
                       int Kd, int Ksrc, int Nw) {
  int idx = blockIdx.x * blockDim.x + threadIdx.x;
  if (idx >= Kd * Nw) return;
  int h    = idx & 15;
  int lane = (idx >> 4) & 31;
  int tile = idx >> 9;
  int ntiles = Nw >> 4;
  int kt = tile / ntiles, nt = tile - kt * ntiles;
  int n  = nt * 16 + (lane & 15);
  int kb = (lane < 16) ? 0 : 8;
  int k  = kt * 32 + kb + (h & 7) + ((h >> 3) << 4);
  float v = (k < Ksrc) ? W[(size_t)k * Nw + n] : 0.0f;
  P[idx] = (_Float16)v;
}

// ---------------- small utility kernels --------------------------------------

// t (B,512,C) -> row-major transposed f16 (N=8192, 512)
__global__ void k_tT(const float* __restrict__ T, _Float16* __restrict__ O) {
  int idx = blockIdx.x * blockDim.x + threadIdx.x;
  if (idx >= NROWS * 512) return;
  int r = idx >> 9, k = idx & 511;
  int b = r >> 7, c = r & 127;
  O[idx] = (_Float16)T[((size_t)b * 512 + k) * 128 + c];
}

// final (N,96) -> (B,96,C)
__global__ void k_out(const float* __restrict__ R, float* __restrict__ O) {
  int idx = blockIdx.x * blockDim.x + threadIdx.x;
  if (idx >= NROWS * 96) return;
  int r = idx / 96, p = idx % 96;
  int b = r >> 7, c = r & 127;
  O[((size_t)b * 96 + p) * 128 + c] = R[idx];
}

// ---------------- DTW + clustering + embed (one block per row) ---------------

__device__ inline float dtw16(const float* __restrict__ a, const float* __restrict__ b) {
  float prev[16];
  float acc = 0.f;
#pragma unroll
  for (int j = 0; j < 16; ++j) { acc += fabsf(a[0] - b[j]); prev[j] = acc; }
#pragma unroll
  for (int i = 1; i < 16; ++i) {
    float ai = a[i];
    float left = BIGF, pl = BIGF;
#pragma unroll
    for (int j = 0; j < 16; ++j) {
      float pj = prev[j];
      float m  = fminf(pj, pl);
      float d  = fabsf(ai - b[j]) + fminf(left, m);
      prev[j] = d; pl = pj; left = d;
    }
  }
  return prev[15];
}

__global__ __launch_bounds__(64) void k_reps(
    const float* __restrict__ S, const float* __restrict__ Wemb,
    const float* __restrict__ bemb, float* __restrict__ X) {
  __shared__ float sS[520];
  __shared__ float sD[63];
  __shared__ float sDc[64];
  __shared__ float sCent[3 * 16];
  __shared__ int   sGid[64];
  __shared__ int   sCnt[3];
  __shared__ int   sRep[3];
  int tid = threadIdx.x;
  int r = blockIdx.x;
  int b = r >> 7, c = r & 127;
  for (int i = tid; i < 512; i += 64) sS[i] = S[((size_t)b * 512 + i) * 128 + c];
  __syncthreads();
  if (tid < 8) sS[512 + tid] = sS[511];
  __syncthreads();
  if (tid >= 1) {                       // d_cons[tid-1] = dtw(patch[tid], patch[tid-1])
    float bb[16];
#pragma unroll
    for (int j = 0; j < 16; ++j) bb[j] = sS[(tid - 1) * 8 + j];
    sD[tid - 1] = dtw16(&sS[tid * 8], bb);
  }
  __syncthreads();
  if (tid == 0) {                       // top-2 -> change points -> group ids
    int i1 = 0; float v1 = sD[0];
    for (int q = 1; q < 63; ++q) if (sD[q] > v1) { v1 = sD[q]; i1 = q; }
    int i2 = (i1 == 0) ? 1 : 0; float v2 = sD[i2];
    for (int q = 0; q < 63; ++q) if (q != i1 && sD[q] > v2) { v2 = sD[q]; i2 = q; }
    int s0 = ((i1 < i2) ? i1 : i2) + 1;
    int s1 = ((i1 < i2) ? i2 : i1) + 1;
    int cnt0 = 0, cnt1 = 0, cnt2 = 0;
    for (int p = 0; p < 64; ++p) {
      int g = (p >= s0) + (p >= s1);
      sGid[p] = g;
      if (g == 0) ++cnt0; else if (g == 1) ++cnt1; else ++cnt2;
    }
    sCnt[0] = cnt0; sCnt[1] = cnt1; sCnt[2] = cnt2;
  }
  __syncthreads();
  if (tid < 48) {                       // centroids
    int k = tid >> 4, j = tid & 15;
    float sum = 0.f;
    for (int p = 0; p < 64; ++p) if (sGid[p] == k) sum += sS[p * 8 + j];
    sCent[k * 16 + j] = sum / (float)sCnt[k];
  }
  __syncthreads();
  {                                     // d_cent
    float bb[16];
    int g = sGid[tid];
#pragma unroll
    for (int j = 0; j < 16; ++j) bb[j] = sCent[g * 16 + j];
    sDc[tid] = dtw16(&sS[tid * 8], bb);
  }
  __syncthreads();
  if (tid < 3) {                        // per-group argmin -> representative
    float best = 2.0f * BIGF; int bi = 0;
    for (int p = 0; p < 64; ++p)
      if (sGid[p] == tid && sDc[p] < best) { best = sDc[p]; bi = p; }
    sRep[tid] = bi;
  }
  __syncthreads();
  {                                     // x[r][k][:] = patch[rep] @ Wemb + bemb
    int d = tid;
    for (int k = 0; k < 3; ++k) {
      int rp = sRep[k];
      float acc = bemb[d];
#pragma unroll
      for (int j = 0; j < 16; ++j) acc += sS[rp * 8 + j] * Wemb[j * 64 + d];
      X[((size_t)r * 3 + k) * 64 + d] = acc;
    }
  }
}

// ---------------- fused transformer encoder layer ----------------------------
// 1 block = 16 rows = 48 tokens (3 M-tiles), 4 waves. All weights pre-packed.

__global__ __launch_bounds__(128) void k_encoder(
    float* __restrict__ X, _Float16* __restrict__ XH,
    const _Float16* __restrict__ Wqkv, const float* __restrict__ bqkv,
    const _Float16* __restrict__ Wo,   const float* __restrict__ bo,
    const _Float16* __restrict__ W1,   const float* __restrict__ bf1,
    const _Float16* __restrict__ W2,   const float* __restrict__ bf2,
    const float* __restrict__ g1, const float* __restrict__ b1,
    const float* __restrict__ g2, const float* __restrict__ b2) {
  __shared__ __align__(16) unsigned char smem[49152];
  float*    sX   = (float*)smem;                // 48*64 f32   (x / residual / x1)
  _Float16* sA   = (_Float16*)(smem + 12288);   // 48*64 f16   (A operand staging)
  _Float16* sQKV = (_Float16*)(smem + 18432);   // 48*192 f16
  float*    sO   = (float*)(smem + 36864);      // 48*64 f32   (attention out)
  _Float16* sH   = (_Float16*)(smem + 18432);   // 48*256 f16  (FF chunk, aliases QKV+O)

  int tid = threadIdx.x, lane = tid & 31, wv = tid >> 5;
  size_t t0 = (size_t)blockIdx.x * 48;

  for (int i = tid; i < 48 * 64; i += 128) {
    float v = X[t0 * 64 + i];
    sX[i] = v; sA[i] = (_Float16)v;
  }
  __syncthreads();

  // QKV: 48x192 = 3x12 tiles, K=64
  for (int tix = wv; tix < 36; tix += 4) {
    int mt = tix % 3, nt = tix / 3;
    v8f acc = vzero8();
    for (int kt = 0; kt < 2; ++kt) {
      v16h a = frag_a(sA, 64, mt * 16, kt * 32, lane);
      v16h b = frag_bp(Wqkv, kt, nt, 12, lane);
      acc = wmma32f16(a, b, acc);
    }
    int n = nt * 16 + (lane & 15);
    float bb = bqkv[n];
    int mo = (lane >= 16) ? 8 : 0;
#pragma unroll
    for (int i = 0; i < 8; ++i) {
      int m = mt * 16 + i + mo;
      sQKV[m * 192 + n] = (_Float16)(acc[i] + bb);
    }
  }
  __syncthreads();

  // attention: (row, head) per thread; S=3, dh=16
  if (tid < 64) {
    int row = tid >> 2, h = tid & 3;
    float att[3][3];
    for (int i = 0; i < 3; ++i)
      for (int j = 0; j < 3; ++j) {
        float d = 0.f;
        for (int e = 0; e < 16; ++e)
          d += (float)sQKV[(row * 3 + i) * 192 + h * 16 + e] *
               (float)sQKV[(row * 3 + j) * 192 + 64 + h * 16 + e];
        att[i][j] = d * 0.25f;
      }
    for (int i = 0; i < 3; ++i) {
      float mx = fmaxf(att[i][0], fmaxf(att[i][1], att[i][2]));
      float e0 = expf(att[i][0] - mx), e1 = expf(att[i][1] - mx), e2 = expf(att[i][2] - mx);
      float inv = 1.0f / (e0 + e1 + e2);
      att[i][0] = e0 * inv; att[i][1] = e1 * inv; att[i][2] = e2 * inv;
    }
    for (int i = 0; i < 3; ++i)
      for (int e = 0; e < 16; ++e) {
        float o = 0.f;
        for (int j = 0; j < 3; ++j)
          o += att[i][j] * (float)sQKV[(row * 3 + j) * 192 + 128 + h * 16 + e];
        sO[(row * 3 + i) * 64 + h * 16 + e] = o;
      }
  }
  __syncthreads();
  for (int i = tid; i < 48 * 64; i += 128) sA[i] = (_Float16)sO[i];
  __syncthreads();

  // O-projection + residual: wave wv owns N-tile wv
  for (int mt = 0; mt < 3; ++mt) {
    v8f acc = vzero8();
    for (int kt = 0; kt < 2; ++kt) {
      v16h a = frag_a(sA, 64, mt * 16, kt * 32, lane);
      v16h b = frag_bp(Wo, kt, wv, 4, lane);
      acc = wmma32f16(a, b, acc);
    }
    int n = wv * 16 + (lane & 15);
    float bb = bo[n];
    int mo = (lane >= 16) ? 8 : 0;
#pragma unroll
    for (int i = 0; i < 8; ++i) {
      int m = mt * 16 + i + mo;
      sX[m * 64 + n] += acc[i] + bb;
    }
  }
  __syncthreads();

  // LN1 (per token)
  if (tid < 48) {
    float mu = 0.f;
    for (int d = 0; d < 64; ++d) mu += sX[tid * 64 + d];
    mu *= (1.0f / 64.0f);
    float var = 0.f;
    for (int d = 0; d < 64; ++d) { float dv = sX[tid * 64 + d] - mu; var += dv * dv; }
    var *= (1.0f / 64.0f);
    float rs = rsqrtf(var + 1e-5f);
    for (int d = 0; d < 64; ++d) {
      float y = (sX[tid * 64 + d] - mu) * rs * g1[d] + b1[d];
      sX[tid * 64 + d] = y; sA[tid * 64 + d] = (_Float16)y;
    }
  }
  __syncthreads();

  // FF 64->2048->64 fused, hidden chunked by 256 columns in LDS
  v8f facc[3] = {vzero8(), vzero8(), vzero8()};
  for (int ch = 0; ch < 8; ++ch) {
    for (int j = 0; j < 4; ++j) {
      int nl = wv * 4 + j;
      for (int mt = 0; mt < 3; ++mt) {
        v8f acc = vzero8();
        for (int kt = 0; kt < 2; ++kt) {
          v16h a = frag_a(sA, 64, mt * 16, kt * 32, lane);
          v16h b = frag_bp(W1, kt, ch * 16 + nl, 128, lane);
          acc = wmma32f16(a, b, acc);
        }
        int nloc = nl * 16 + (lane & 15);
        float bb = bf1[ch * 256 + nloc];
        int mo = (lane >= 16) ? 8 : 0;
#pragma unroll
        for (int i = 0; i < 8; ++i) {
          float v = acc[i] + bb; v = v > 0.f ? v : 0.f;   // relu
          int m = mt * 16 + i + mo;
          sH[m * 256 + nloc] = (_Float16)v;
        }
      }
    }
    __syncthreads();
    for (int ks = 0; ks < 8; ++ks) {
      v16h b = frag_bp(W2, ch * 8 + ks, wv, 4, lane);
      for (int mt = 0; mt < 3; ++mt) {
        v16h a = frag_a(sH, 256, mt * 16, ks * 32, lane);
        facc[mt] = wmma32f16(a, b, facc[mt]);
      }
    }
    __syncthreads();
  }
  {
    int n = wv * 16 + (lane & 15);
    float bb = bf2[n];
    int mo = (lane >= 16) ? 8 : 0;
    for (int mt = 0; mt < 3; ++mt)
#pragma unroll
      for (int i = 0; i < 8; ++i) {
        int m = mt * 16 + i + mo;
        sX[m * 64 + n] += facc[mt][i] + bb;
      }
  }
  __syncthreads();

  // LN2 + writeback (f32 + f16 copy for downstream GEMMs)
  if (tid < 48) {
    float mu = 0.f;
    for (int d = 0; d < 64; ++d) mu += sX[tid * 64 + d];
    mu *= (1.0f / 64.0f);
    float var = 0.f;
    for (int d = 0; d < 64; ++d) { float dv = sX[tid * 64 + d] - mu; var += dv * dv; }
    var *= (1.0f / 64.0f);
    float rs = rsqrtf(var + 1e-5f);
    for (int d = 0; d < 64; ++d) {
      float y = (sX[tid * 64 + d] - mu) * rs * g2[d] + b2[d];
      X[(t0 + tid) * 64 + d] = y;
      XH[(t0 + tid) * 64 + d] = (_Float16)y;
    }
  }
}

// ---------------- generic WMMA GEMM: out = act(A @ Wpacked + bias) -----------
// A: M x Kd f16 (row-major, lda); W: packed fragment layout (Kd/32 x Ntiles).
// grid.x = M/16, grid.y = ceil(Ntiles/4); wave per N-tile.

__global__ __launch_bounds__(128) void k_gemm(
    const _Float16* __restrict__ A, int lda,
    const _Float16* __restrict__ W,
    const float* __restrict__ bias,
    float* __restrict__ o32, int ldo32,
    _Float16* __restrict__ o16, int ldo16,
    int Kd, int Ntiles, int act) {
  int lane = threadIdx.x & 31;
  int wv   = threadIdx.x >> 5;
  int m0 = blockIdx.x * 16;
  int nt = blockIdx.y * 4 + wv;
  if (nt >= Ntiles) return;
  int n0 = nt * 16;
  v8f acc = vzero8();
  for (int k0 = 0; k0 < Kd; k0 += 32) {
    v16h a = frag_a(A, lda, m0, k0, lane);
    v16h b = frag_bp(W, k0 >> 5, nt, Ntiles, lane);
    acc = wmma32f16(a, b, acc);
  }
  int n = n0 + (lane & 15);
  float bb = bias ? bias[n] : 0.0f;
  int mo = (lane >= 16) ? 8 : 0;
#pragma unroll
  for (int i = 0; i < 8; ++i) {
    int m = m0 + i + mo;
    float v = acc[i] + bb;
    if (act == 1) v = 0.5f * v * (1.0f + erff(v * 0.70710678118f));  // exact gelu
    if (o32) o32[(size_t)m * ldo32 + n] = v;
    if (o16) o16[(size_t)m * ldo16 + n] = (_Float16)v;
  }
}

// ---------------- pair-mean + layernorm (one wave per row) -------------------

__global__ __launch_bounds__(32) void k_pmln(
    const float* __restrict__ in, int L,
    const float* __restrict__ g, const float* __restrict__ bta,
    _Float16* __restrict__ o16, int ldo, int padTo) {
  int r = blockIdx.x, lane = threadIdx.x;
  const float* row = in + (size_t)r * (2 * L);
  float y[6];
  int per = (L + 31) / 32;
  float sum = 0.f;
  for (int u = 0; u < per; ++u) {
    int l = lane + u * 32;
    float v = 0.f;
    if (l < L) v = (row[2 * l] + row[2 * l + 1]) * 0.5f;
    y[u] = v; sum += v;
  }
  for (int o = 16; o > 0; o >>= 1) sum += __shfl_down(sum, o, 32);
  sum = __shfl(sum, 0, 32);
  float mu = sum / (float)L;
  float sq = 0.f;
  for (int u = 0; u < per; ++u) {
    int l = lane + u * 32;
    if (l < L) { float dv = y[u] - mu; sq += dv * dv; }
  }
  for (int o = 16; o > 0; o >>= 1) sq += __shfl_down(sq, o, 32);
  sq = __shfl(sq, 0, 32);
  float rs = rsqrtf(sq / (float)L + 1e-5f);
  for (int u = 0; u < per; ++u) {
    int l = lane + u * 32;
    if (l < L) o16[(size_t)r * ldo + l] = (_Float16)((y[u] - mu) * rs * g[l] + bta[l]);
  }
  for (int l = L + lane; l < padTo; l += 32) o16[(size_t)r * ldo + l] = (_Float16)0.f;
}

// ---------------- launch -----------------------------------------------------

extern "C" void kernel_launch(void* const* d_in, const int* in_sizes, int n_in,
                              void* d_out, int out_size, void* d_ws, size_t ws_size,
                              hipStream_t stream) {
  (void)in_sizes; (void)n_in; (void)out_size; (void)ws_size;
  auto in = [&](int i) { return (const float*)d_in[i]; };
  const float* S = in(0);
  const float* T = in(1);
  const float* Wemb = in(2); const float* bemb = in(3);
  const float* bout1 = in(29);
  const float* bout2 = in(31);
  const float* bt1 = in(33);
  const float* glt1 = in(34); const float* blt1 = in(35);
  const float* bt2 = in(37);
  const float* glt2 = in(38); const float* blt2 = in(39);
  const float* bt3 = in(41);
  const float* bcat = in(43);

  unsigned char* ws = (unsigned char*)d_ws;
  size_t off = 0;
  auto alloc = [&](size_t bytes) -> void* {
    void* p = ws + off;
    off += (bytes + 255) & ~(size_t)255;
    return p;
  };

  const int N = NROWS;
  float*    xf    = (float*)alloc((size_t)N * 3 * 64 * 4);
  _Float16* xh    = (_Float16*)alloc((size_t)N * 192 * 2);
  _Float16* h16   = (_Float16*)alloc((size_t)N * 192 * 2);
  _Float16* cat16 = (_Float16*)alloc((size_t)N * 192 * 2);
  float*    resf  = (float*)alloc((size_t)N * 96 * 4);
  _Float16* tT16  = (_Float16*)alloc((size_t)N * 512 * 2);
  float*    tt1   = (float*)alloc((size_t)N * 384 * 4);
  _Float16* ttA   = (_Float16*)alloc((size_t)N * 192 * 2);
  float*    tt2   = (float*)alloc((size_t)N * 96 * 4);
  _Float16* ttB   = (_Float16*)alloc((size_t)N * 64 * 2);
  _Float16* wqkv16[2]; _Float16* wo16[2]; _Float16* w116[2]; _Float16* w216[2];
  for (int l = 0; l < 2; ++l) {
    wqkv16[l] = (_Float16*)alloc(12288 * 2);
    wo16[l]   = (_Float16*)alloc(4096 * 2);
    w116[l]   = (_Float16*)alloc(131072 * 2);
    w216[l]   = (_Float16*)alloc(131072 * 2);
  }
  _Float16* wout116 = (_Float16*)alloc(36864 * 2);
  _Float16* wout216 = (_Float16*)alloc(18432 * 2);
  _Float16* wt116   = (_Float16*)alloc(196608 * 2);
  _Float16* wt216   = (_Float16*)alloc(18432 * 2);
  _Float16* wt316   = (_Float16*)alloc(6144 * 2);
  _Float16* wcat16  = (_Float16*)alloc(18432 * 2);

  auto pack = [&](const float* src, _Float16* dst, int Kd, int Ksrc, int Nw) {
    int n = Kd * Nw;
    k_pack<<<(n + 255) / 256, 256, 0, stream>>>(src, dst, Kd, Ksrc, Nw);
  };
  for (int l = 0; l < 2; ++l) {
    int base = 4 + l * 12;  // Wqkv,bqkv,Wo,bo,W1,bf1,W2,bf2,g1,b1,g2,b2
    pack(in(base + 0), wqkv16[l], 64, 64, 192);
    pack(in(base + 2), wo16[l],   64, 64, 64);
    pack(in(base + 4), w116[l],   64, 64, 2048);
    pack(in(base + 6), w216[l],   2048, 2048, 64);
  }
  pack(in(28), wout116, 192, 192, 192);
  pack(in(30), wout216, 192, 192, 96);
  pack(in(32), wt116,   512, 512, 384);
  pack(in(36), wt216,   192, 192, 96);
  pack(in(40), wt316,   64, 48, 96);     // K padded 48 -> 64 with zeros
  pack(in(42), wcat16,  192, 192, 96);
  k_tT<<<(N * 512 + 255) / 256, 256, 0, stream>>>(T, tT16);

  // DTW clustering + representative embedding -> x (N,3,64)
  k_reps<<<N, 64, 0, stream>>>(S, Wemb, bemb, xf);

  // two fused encoder layers
  for (int l = 0; l < 2; ++l) {
    int base = 4 + l * 12;
    k_encoder<<<N / 16, 128, 0, stream>>>(
        xf, xh, wqkv16[l], in(base + 1), wo16[l], in(base + 3), w116[l],
        in(base + 5), w216[l], in(base + 7), in(base + 8), in(base + 9),
        in(base + 10), in(base + 11));
  }

  // s-branch head: h = gelu(x_flat @ W_out1 + b); s_out = h @ W_out2 + b -> cat[:, :96]
  k_gemm<<<dim3(N / 16, 3), 128, 0, stream>>>(xh, 192, wout116, bout1,
                                              nullptr, 0, h16, 192, 192, 12, 1);
  k_gemm<<<dim3(N / 16, 2), 128, 0, stream>>>(h16, 192, wout216, bout2,
                                              nullptr, 0, cat16, 192, 192, 6, 0);

  // t-branch
  k_gemm<<<dim3(N / 16, 6), 128, 0, stream>>>(tT16, 512, wt116, bt1,
                                              tt1, 384, nullptr, 0, 512, 24, 0);
  k_pmln<<<N, 32, 0, stream>>>(tt1, 192, glt1, blt1, ttA, 192, 192);
  k_gemm<<<dim3(N / 16, 2), 128, 0, stream>>>(ttA, 192, wt216, bt2,
                                              tt2, 96, nullptr, 0, 192, 6, 0);
  k_pmln<<<N, 32, 0, stream>>>(tt2, 48, glt2, blt2, ttB, 64, 64);
  k_gemm<<<dim3(N / 16, 2), 128, 0, stream>>>(ttB, 64, wt316, bt3,
                                              nullptr, 0, cat16 + 96, 192, 64, 6, 0);

  // concat @ W_cat + b -> (N,96), then transpose to (B,96,C)
  k_gemm<<<dim3(N / 16, 2), 128, 0, stream>>>(cat16, 192, wcat16, bcat,
                                              resf, 96, nullptr, 0, 192, 6, 0);
  k_out<<<(N * 96 + 255) / 256, 256, 0, stream>>>(resf, (float*)d_out);
}
